// AUGRU_88751204204934
// MI455X (gfx1250) — compile-verified, matrix operand order checked by reading, not run on hardware
//
#include <hip/hip_runtime.h>
#include <hip/hip_bf16.h>
#include <math.h>

#define B_ 4096
#define T_ 200
#define E_ 64

typedef __attribute__((ext_vector_type(16))) _Float16 v16h;
typedef __attribute__((ext_vector_type(8)))  _Float16 v8h;
typedef __attribute__((ext_vector_type(4)))  _Float16 v4h;
typedef __attribute__((ext_vector_type(8)))  float    v8f;

#define WMMA_F16(a, b, c) \
  __builtin_amdgcn_wmma_f32_16x16x32_f16(false, (a), false, (b), (short)0, (c), false, false)

#define LDS_FENCE() do { asm volatile("s_wait_dscnt 0" ::: "memory"); \
                         __builtin_amdgcn_wave_barrier(); } while (0)

// fast device math: v_exp_f32 / v_rcp_f32 based
__device__ __forceinline__ float fast_sigmoid(float x) {
  return __builtin_amdgcn_rcpf(1.0f + __expf(-x));
}
__device__ __forceinline__ float fast_tanh(float x) {
  return 1.0f - 2.0f * __builtin_amdgcn_rcpf(__expf(2.0f * x) + 1.0f);
}

// A fragment (16-bit 16x32 layout): lane (m = L%16, half = L/16) holds two
// contiguous 8-element runs: K = k0+8*half+0..7 and K = k0+16+8*half+0..7.
__device__ __forceinline__ v16h ld_frag_a(const _Float16* buf, int m, int half, int ld, int k0) {
  v8h lo = *(const v8h*)(buf + m * ld + k0 + 8 * half);
  v8h hi = *(const v8h*)(buf + m * ld + k0 + 16 + 8 * half);
  return __builtin_shufflevector(lo, hi, 0, 1, 2, 3, 4, 5, 6, 7, 8, 9, 10, 11, 12, 13, 14, 15);
}

// B fragment from a pre-swizzled buffer: frag-major, 32B contiguous per lane.
__device__ __forceinline__ v16h ld_frag_s(const _Float16* w, int frag, int lane) {
  return *(const v16h*)(w + frag * 512 + lane * 16);
}

// ============================================================================
// Kernel 1: collapse the affine attention MLP.
//   w1f[0..255] = W1 @ (W2 @ Wf);  w1f[256] = b1.(W2@Wf) + b2.Wf + bf
// ============================================================================
__global__ void collapse_kernel(const float* __restrict__ W1, const float* __restrict__ b1,
                                const float* __restrict__ W2, const float* __restrict__ b2,
                                const float* __restrict__ Wf, const float* __restrict__ bf,
                                float* __restrict__ w1f) {
  __shared__ float w2f[80];
  int tid = threadIdx.x;
  if (tid < 80) {
    float s = 0.0f;
#pragma unroll 8
    for (int j = 0; j < 40; ++j) s += W2[tid * 40 + j] * Wf[j];
    w2f[tid] = s;
  }
  __syncthreads();
  {
    float s = 0.0f;
#pragma unroll 8
    for (int k = 0; k < 80; ++k) s += W1[tid * 80 + k] * w2f[k];
    w1f[tid] = s;
  }
  if (tid == 0) {
    float c = bf[0];
    for (int k = 0; k < 80; ++k) c += b1[k] * w2f[k];
    for (int j = 0; j < 40; ++j) c += b2[j] * Wf[j];
    w1f[256] = c;
  }
}

// ============================================================================
// Kernel 2: streaming attention logits (HBM-bound: reads gru exactly once).
//   logit(b,t) = k(b,t) . [ (wB - wC) + q(b) * wD ] + q(b).(wA + wC) + c
//   One block per batch row b; threads t=0..199 each do a 64-float dot.
//   Masked (==0 -> -inf) and stored transposed [T][B] for the softmax/AUGRU.
// ============================================================================
__global__ void attn_logits_kernel(const float* __restrict__ gru, const float* __restrict__ tgt,
                                   const float* __restrict__ w1f, float* __restrict__ logits) {
  __shared__ float qwS[64];
  __shared__ float red[64];
  __shared__ float qcS;
  const int b = blockIdx.x, tid = threadIdx.x;
  if (tid < 64) {
    float q  = tgt[b * 64 + tid];
    float wA = w1f[tid], wB = w1f[64 + tid], wC = w1f[128 + tid], wD = w1f[192 + tid];
    qwS[tid] = (wB - wC) + q * wD;
    red[tid] = q * (wA + wC);
  }
  __syncthreads();
  if (tid < 32) red[tid] += red[tid + 32];
  __syncthreads();
  if (tid < 16) red[tid] += red[tid + 16];
  __syncthreads();
  if (tid < 8)  red[tid] += red[tid + 8];
  __syncthreads();
  if (tid < 4)  red[tid] += red[tid + 4];
  __syncthreads();
  if (tid < 2)  red[tid] += red[tid + 2];
  __syncthreads();
  if (tid == 0) qcS = red[0] + red[1] + w1f[256];
  __syncthreads();

  if (tid < T_) {
    const float4* row = (const float4*)(gru + ((size_t)b * T_ + tid) * E_);
    float d = 0.0f;
#pragma unroll
    for (int j = 0; j < 16; ++j) {
      float4 xv = row[j];
      d += xv.x * qwS[j * 4] + xv.y * qwS[j * 4 + 1] + xv.z * qwS[j * 4 + 2] + xv.w * qwS[j * 4 + 3];
    }
    float lg = d + qcS;
    if (lg == 0.0f) lg = -__builtin_inff();
    logits[(size_t)tid * B_ + b] = lg;
  }
}

// ============================================================================
// Kernel 3: softmax over the batch axis, per timestep (contiguous columns)
// ============================================================================
__global__ void softmax_col_kernel(const float* __restrict__ logits, float* __restrict__ scores) {
  __shared__ float red[256];
  const int t = blockIdx.x, tid = threadIdx.x;
  const float* col = logits + (size_t)t * B_;
  float m = -__builtin_inff();
  for (int i = tid; i < B_; i += 256) m = fmaxf(m, col[i]);
  red[tid] = m; __syncthreads();
  for (int s = 128; s > 0; s >>= 1) { if (tid < s) red[tid] = fmaxf(red[tid], red[tid + s]); __syncthreads(); }
  m = red[0]; __syncthreads();
  float sum = 0.0f;
  for (int i = tid; i < B_; i += 256) sum += __expf(col[i] - m);
  red[tid] = sum; __syncthreads();
  for (int s = 128; s > 0; s >>= 1) { if (tid < s) red[tid] += red[tid + s]; __syncthreads(); }
  float inv = 1.0f / red[0];
  for (int i = tid; i < B_; i += 256) scores[(size_t)t * B_ + i] = __expf(col[i] - m) * inv;
}

// ============================================================================
// Kernel 4: AUGRU recurrence. One wave owns 16 batch rows for all 200 steps.
//   r is scalar per row => (h*r)@Wch = r*(h@Wch). ONE staging of [x,h] per
//   step feeds: gate WMMA (u,r dots vs [Wu|Wr|0..]) and candidate WMMAs
//   split into P = x@Wcx and Q = h@Wch (both init from inline 0; bias added
//   as a scalar in the activation). h = (1-u)h + u*tanh(P + r*Q + bc).
//   20 v_wmma / step, h in registers, wave-local LDS fences only.
// ============================================================================
__global__ void __launch_bounds__(128, 1)
augru_kernel(const float* __restrict__ gru, const float* __restrict__ scores,
             const float* __restrict__ Wu, const float* __restrict__ bu,
             const float* __restrict__ Wr, const float* __restrict__ br,
             const float* __restrict__ Wc, const float* __restrict__ bc,
             float* __restrict__ out) {
  extern __shared__ char smem[];
  _Float16* WcS = (_Float16*)smem;                  // 16 frags * 512 = 16384B
  _Float16* WgS = (_Float16*)(smem + 16384);        // 4 frags * 512  = 4096B
  float*    bcs = (float*)(smem + 16384 + 4096);    // [64]
  char*   wbase = smem + 16384 + 4096 + 256;        // per-wave: cat 4096 + u/r 128 = 4224B

  const int tid = threadIdx.x, wave = tid >> 5, lane = tid & 31;
  const int nIdx = lane & 15, half = lane >> 4;

  // Stage Wc in fragment order (kc 0,1 -> x-part Wcx; kc 2,3 -> h-part Wch)
  for (int idx = tid; idx < 16 * 512; idx += 128) {
    int frag = idx >> 9, within = idx & 511;
    int l = within >> 4, e = within & 15;
    int nt = frag >> 2, kc = frag & 3;
    int n = nt * 16 + (l & 15);
    int k = kc * 32 + ((e < 8) ? 0 : 16) + 8 * (l >> 4) + (e & 7);
    WcS[idx] = (_Float16)Wc[k * 64 + n];
  }
  // Stage gate matrix [Wu | Wr | 0...] (128x16) in fragment order
  for (int idx = tid; idx < 4 * 512; idx += 128) {
    int frag = idx >> 9, within = idx & 511;
    int l = within >> 4, e = within & 15;
    int n = l & 15;
    int k = frag * 32 + ((e < 8) ? 0 : 16) + 8 * (l >> 4) + (e & 7);
    float v = (n == 0) ? Wu[k] : (n == 1) ? Wr[k] : 0.0f;
    WgS[idx] = (_Float16)v;
  }
  if (tid < 64) bcs[tid] = bc[tid];
  __syncthreads();

  _Float16* cat = (_Float16*)(wbase + wave * 4224);      // [16][128] = [x_t , h]
  float* uArr = (float*)(wbase + wave * 4224 + 4096);    // [16]
  float* rArr = uArr + 16;                               // [16]

  const int base = (blockIdx.x * 4 + wave) * 16;
  const int brow = base + nIdx;  // batch row this lane stages x for
  const float buv = bu[0], brv = br[0];
  float bcv[4];
#pragma unroll
  for (int nt = 0; nt < 4; ++nt) bcv[nt] = bcs[nt * 16 + nIdx];

  v8f h[4] = {};  // hidden state, accumulator layout: h[nt][v] = h[v+8*half][nt*16+nIdx]

  for (int t = 0; t < T_; ++t) {
    // 1) stage x_t into cat[:, 0:64] (packed 8B stores) + prefetch next step
    const float4* xrow = (const float4*)(gru + ((size_t)brow * T_ + t) * E_ + half * 32);
    if (t + 1 < T_) __builtin_prefetch(gru + ((size_t)brow * T_ + t + 1) * E_ + half * 32, 0, 3);
#pragma unroll
    for (int j = 0; j < 8; ++j) {
      float4 xv = xrow[j];
      v4h xh = { (_Float16)xv.x, (_Float16)xv.y, (_Float16)xv.z, (_Float16)xv.w };
      *(v4h*)(cat + nIdx * 128 + half * 32 + j * 4) = xh;
    }
    // 2) stage h (f16) into cat[:, 64:128] from registers
#pragma unroll
    for (int nt = 0; nt < 4; ++nt)
#pragma unroll
      for (int v = 0; v < 8; ++v)
        cat[(v + 8 * half) * 128 + 64 + nt * 16 + nIdx] = (_Float16)h[nt][v];
    LDS_FENCE();

    // 3) one set of A fragments feeds gate AND candidate WMMAs
    v16h a0 = ld_frag_a(cat, nIdx, half, 128, 0);
    v16h a1 = ld_frag_a(cat, nIdx, half, 128, 32);
    v16h a2 = ld_frag_a(cat, nIdx, half, 128, 64);
    v16h a3 = ld_frag_a(cat, nIdx, half, 128, 96);

    v8f g = {};  // col0 = u-dot, col1 = r-dot
    g = WMMA_F16(a0, ld_frag_s(WgS, 0, lane), g);
    g = WMMA_F16(a1, ld_frag_s(WgS, 1, lane), g);
    g = WMMA_F16(a2, ld_frag_s(WgS, 2, lane), g);
    g = WMMA_F16(a3, ld_frag_s(WgS, 3, lane), g);

    v8f accP[4], accQ[4];  // P = x@Wcx ; Q = h@Wch  (both start from inline 0)
#pragma unroll
    for (int nt = 0; nt < 4; ++nt) {
      v8f p = {}; v8f q = {};
      p = WMMA_F16(a0, ld_frag_s(WcS, nt * 4 + 0, lane), p);
      p = WMMA_F16(a1, ld_frag_s(WcS, nt * 4 + 1, lane), p);
      q = WMMA_F16(a2, ld_frag_s(WcS, nt * 4 + 2, lane), q);
      q = WMMA_F16(a3, ld_frag_s(WcS, nt * 4 + 3, lane), q);
      accP[nt] = p; accQ[nt] = q;
    }

    // 4) distribute gates: lane col0 -> u (with attention score), col1 -> r
    if (nIdx < 2) {
#pragma unroll
      for (int v = 0; v < 8; ++v) {
        int row = v + 8 * half;
        if (nIdx == 0) {
          float si = scores[(size_t)t * B_ + base + row];
          uArr[row] = si * fast_sigmoid(g[v] + buv);
        } else {
          rArr[row] = fast_sigmoid(g[v] + brv);
        }
      }
    }
    LDS_FENCE();
    float u8[8], r8[8];
#pragma unroll
    for (int v = 0; v < 8; ++v) { u8[v] = uArr[v + 8 * half]; r8[v] = rArr[v + 8 * half]; }

    // 5) in-register state update: h = (1-u)h + u*tanh(P + r*Q + bc)
#pragma unroll
    for (int nt = 0; nt < 4; ++nt)
#pragma unroll
      for (int v = 0; v < 8; ++v) {
        float tilde = fast_tanh(accP[nt][v] + r8[v] * accQ[nt][v] + bcv[nt]);
        h[nt][v] = (1.0f - u8[v]) * h[nt][v] + u8[v] * tilde;
      }
  }

  // ---- write hT (accumulator-layout scatter) ----
#pragma unroll
  for (int nt = 0; nt < 4; ++nt)
#pragma unroll
    for (int v = 0; v < 8; ++v)
      out[(size_t)(base + v + 8 * half) * 64 + nt * 16 + nIdx] = h[nt][v];
}

// ============================================================================
extern "C" void kernel_launch(void* const* d_in, const int* in_sizes, int n_in,
                              void* d_out, int out_size, void* d_ws, size_t ws_size,
                              hipStream_t stream) {
  const float* gru = (const float*)d_in[0];
  const float* tgt = (const float*)d_in[1];
  const float* W1  = (const float*)d_in[2];
  const float* b1  = (const float*)d_in[3];
  const float* W2  = (const float*)d_in[4];
  const float* b2  = (const float*)d_in[5];
  const float* Wf  = (const float*)d_in[6];
  const float* bf  = (const float*)d_in[7];
  const float* Wu  = (const float*)d_in[8];
  const float* bu  = (const float*)d_in[9];
  const float* Wr  = (const float*)d_in[10];
  const float* br  = (const float*)d_in[11];
  const float* Wc  = (const float*)d_in[12];
  const float* bc  = (const float*)d_in[13];
  float* out = (float*)d_out;

  // Workspace layout
  float* w1f    = (float*)d_ws;                                  // 257 floats (+pad)
  float* logits = (float*)((char*)d_ws + 4096);                  // T*B
  float* scores = logits + (size_t)T_ * B_;                      // T*B

  // 1) collapse the affine MLP to a single 256-vector + scalar
  collapse_kernel<<<1, 256, 0, stream>>>(W1, b1, W2, b2, Wf, bf, w1f);

  // 2) streaming logits: one block per batch row, gru read once, coalesced
  attn_logits_kernel<<<B_, 256, 0, stream>>>(gru, tgt, w1f, logits);

  // 3) softmax over the batch axis per timestep
  softmax_col_kernel<<<T_, 256, 0, stream>>>(logits, scores);

  // 4) AUGRU recurrence: 256 waves x 16 rows, 4 waves/block, 20 WMMA/step
  const uint32_t smemC = 16384 + 4096 + 256 + 4 * 4224;          // 37,632 B
  augru_kernel<<<B_ / 64, 128, smemC, stream>>>(gru, scores, Wu, bu, Wr, br, Wc, bc, out);
}
